// AttentionDS_16887811408512
// MI455X (gfx1250) — compile-verified, matrix operand order checked by reading, not run on hardware
//
#include <hip/hip_runtime.h>
#include <math.h>

// ---------------------------------------------------------------------------
// ViT attention block on MI455X (gfx1250, wave32).
// bf16 WMMA (v_wmma_f32_16x16x32_bf16) for all GEMMs, fp32 accumulate.
// Round 2: 32x64 wave tiles in GEMMs, 32-row Q strips in flash attention,
// nontemporal stores for the write-once fp32 outputs.
// ---------------------------------------------------------------------------

#define DIM     768
#define HEADS   12
#define HD      64
#define BATCH   4
#define SEQ     2048
#define MTOT    (BATCH * SEQ)      // 8192
#define QKVN    (3 * DIM)          // 2304
#define SCALEF  0.125f             // 64^-0.5
#define SLICE   ((size_t)MTOT * DIM)   // 6291456 elements per output tensor

typedef __bf16 bf16;
typedef __bf16 v16bf __attribute__((ext_vector_type(16)));
typedef __bf16 v8bf  __attribute__((ext_vector_type(8)));
typedef float  v8f   __attribute__((ext_vector_type(8)));

// Load one A/B fragment for v_wmma_*_16x16x32_bf16.
// Per ISA layout: lane holds row/col (lane&15); K elements at
// kbase + h + {0..7} and kbase + h + 16 + {0..7}, h = (lane>=16) ? 8 : 0.
// Two 16-byte loads (global_load_b128 / ds_load_b128).
__device__ __forceinline__ v16bf load_frag(const bf16* __restrict__ rowptr,
                                           int kbase, int lane) {
  const int h = (lane >> 4) << 3;
  v8bf lo = *(const v8bf*)(rowptr + kbase + h);
  v8bf hi = *(const v8bf*)(rowptr + kbase + h + 16);
  v16bf r;
#pragma unroll
  for (int i = 0; i < 8; ++i) { r[i] = lo[i]; r[i + 8] = hi[i]; }
  return r;
}

__device__ __forceinline__ v8f wmma_bf16(v16bf a, v16bf b, v8f c) {
  return __builtin_amdgcn_wmma_f32_16x16x32_bf16(false, a, false, b,
                                                 (short)0, c, false, false);
}

// ---------------------------------------------------------------------------
// fp32 -> bf16 conversion (grid-stride; NT loads, inputs are read once here)
// ---------------------------------------------------------------------------
__global__ void cvt_f32_bf16(const float* __restrict__ src,
                             bf16* __restrict__ dst, int n) {
  int i = blockIdx.x * blockDim.x + threadIdx.x;
  const int stride = gridDim.x * blockDim.x;
  for (; i < n; i += stride) dst[i] = (bf16)__builtin_nontemporal_load(src + i);
}

// ---------------------------------------------------------------------------
// QKV GEMM: [8192 x 768] @ [2304 x 768]^T.
// Block = 128 thr (4 waves). Block tile 64(M) x 128(N); wave tile 32 x 64.
// Per K-step: 6 fragments (12 b128 loads) feed 8 WMMAs.
// Epilogue: fp32 q/k/v -> d_out slices (NT); bf16 q/k per-head [b,h,n,64],
//           bf16 v transposed per-head [b,h,64,n].
// ---------------------------------------------------------------------------
__global__ __launch_bounds__(128) void qkv_gemm(
    const bf16* __restrict__ xb, const bf16* __restrict__ wb,
    float* __restrict__ qkv_f32,   // points at q slice of d_out; p in {0,1,2}
    bf16* __restrict__ qws, bf16* __restrict__ kws, bf16* __restrict__ vtws) {
  const int lane = threadIdx.x & 31;
  const int wave = threadIdx.x >> 5;
  const int m0 = blockIdx.x * 64 + (wave & 1) * 32;
  const int n0 = blockIdx.y * 128 + (wave >> 1) * 64;

  const bf16* aptr0 = xb + (size_t)(m0 + (lane & 15)) * DIM;
  const bf16* aptr1 = aptr0 + (size_t)16 * DIM;
  const bf16* bptr[4];
#pragma unroll
  for (int t = 0; t < 4; ++t)
    bptr[t] = wb + (size_t)(n0 + t * 16 + (lane & 15)) * DIM;

  v8f acc[2][4];
#pragma unroll
  for (int mt = 0; mt < 2; ++mt)
#pragma unroll
    for (int t = 0; t < 4; ++t) { v8f z = {}; acc[mt][t] = z; }

  for (int k = 0; k < DIM; k += 32) {
    v16bf a0 = load_frag(aptr0, k, lane);
    v16bf a1 = load_frag(aptr1, k, lane);
#pragma unroll
    for (int t = 0; t < 4; ++t) {
      v16bf bfr = load_frag(bptr[t], k, lane);
      acc[0][t] = wmma_bf16(a0, bfr, acc[0][t]);
      acc[1][t] = wmma_bf16(a1, bfr, acc[1][t]);
    }
  }

#pragma unroll
  for (int mt = 0; mt < 2; ++mt) {
    const int mbase = m0 + mt * 16 + ((lane >> 4) << 3);
#pragma unroll
    for (int t = 0; t < 4; ++t) {
      const int d  = n0 + t * 16 + (lane & 15);  // 0..2303
      const int p  = d / DIM;                    // 0=q 1=k 2=v
      const int c  = d % DIM;
      const int hh = c / HD, dd = c % HD;
#pragma unroll
      for (int v = 0; v < 8; ++v) {
        const int m = mbase + v;
        const int b = m >> 11, n = m & (SEQ - 1);
        const float val = acc[mt][t][v];
        __builtin_nontemporal_store(
            val, &qkv_f32[(size_t)p * SLICE + (size_t)m * DIM + c]);
        const bf16 bv = (bf16)val;
        const size_t bh = (size_t)(b * HEADS + hh);
        if (p == 0)      qws [(bh * SEQ + n) * HD + dd]  = bv;
        else if (p == 1) kws [(bh * SEQ + n) * HD + dd]  = bv;
        else             vtws[(bh * HD + dd) * SEQ + n]  = bv;
      }
    }
  }
}

// ---------------------------------------------------------------------------
// Flash attention: softmax(Q K^T * scale) V per (b,h).
// Block = 256 thr (8 waves); wave owns a 32-row Q strip; block covers 256 rows.
// Online softmax over j-blocks of 32 columns. K/V fragments are shared by the
// two M-tiles: 16 b128 loads per 16 WMMAs per j-block.
// ---------------------------------------------------------------------------
__global__ __launch_bounds__(256) void flash_attn(
    const bf16* __restrict__ qws, const bf16* __restrict__ kws,
    const bf16* __restrict__ vtws, bf16* __restrict__ aows) {
  __shared__ bf16 plds[8][32][40];  // per-wave 32x32 P tile, padded rows

  const int lane = threadIdx.x & 31;
  const int wave = threadIdx.x >> 5;
  const int bh = blockIdx.x;                  // 0..47
  const int b = bh / HEADS, h = bh % HEADS;
  const int i0 = blockIdx.y * 256 + wave * 32;

  v16bf qa[2][2];
#pragma unroll
  for (int mt = 0; mt < 2; ++mt) {
    const bf16* qrow =
        qws + ((size_t)bh * SEQ + i0 + mt * 16 + (lane & 15)) * HD;
    qa[mt][0] = load_frag(qrow, 0, lane);
    qa[mt][1] = load_frag(qrow, 32, lane);
  }

  float mrun[2][8], lrun[2][8];
  v8f acc[2][4];
#pragma unroll
  for (int mt = 0; mt < 2; ++mt) {
#pragma unroll
    for (int v = 0; v < 8; ++v) {
      mrun[mt][v] = -__builtin_inff();
      lrun[mt][v] = 0.f;
    }
#pragma unroll
    for (int t = 0; t < 4; ++t) { v8f z = {}; acc[mt][t] = z; }
  }

  const bf16* kbase  = kws  + (size_t)bh * SEQ * HD;
  const bf16* vtbase = vtws + (size_t)bh * HD * SEQ;

  for (int j0 = 0; j0 < SEQ; j0 += 32) {
    // --- K fragments for this 32-column block (shared by both M tiles) ---
    const bf16* krow0 = kbase + (size_t)(j0 + (lane & 15)) * HD;
    const bf16* krow1 = krow0 + (size_t)16 * HD;
    v16bf kb00 = load_frag(krow0, 0, lane);
    v16bf kb01 = load_frag(krow0, 32, lane);
    v16bf kb10 = load_frag(krow1, 0, lane);
    v16bf kb11 = load_frag(krow1, 32, lane);

#pragma unroll
    for (int mt = 0; mt < 2; ++mt) {
      // --- S = Q K^T (16x32 block) ---
      v8f s0 = {}, s1 = {};
      s0 = wmma_bf16(qa[mt][0], kb00, s0);
      s0 = wmma_bf16(qa[mt][1], kb01, s0);
      s1 = wmma_bf16(qa[mt][0], kb10, s1);
      s1 = wmma_bf16(qa[mt][1], kb11, s1);

      // --- online softmax (row stats per acc reg, per 16-lane half) ---
      float p0[8], p1[8], corr[8];
#pragma unroll
      for (int v = 0; v < 8; ++v) {
        const float a = s0[v] * SCALEF, c2 = s1[v] * SCALEF;
        float mx = fmaxf(a, c2);
#pragma unroll
        for (int sh = 1; sh < 16; sh <<= 1)
          mx = fmaxf(mx, __shfl_xor(mx, sh, 32));
        const float nm = fmaxf(mrun[mt][v], mx);
        corr[v] = __expf(mrun[mt][v] - nm);
        mrun[mt][v] = nm;
        p0[v] = __expf(a - nm);
        p1[v] = __expf(c2 - nm);
        float rs = p0[v] + p1[v];
#pragma unroll
        for (int sh = 1; sh < 16; sh <<= 1) rs += __shfl_xor(rs, sh, 32);
        lrun[mt][v] = lrun[mt][v] * corr[v] + rs;
      }
#pragma unroll
      for (int t = 0; t < 4; ++t)
#pragma unroll
        for (int v = 0; v < 8; ++v) acc[mt][t][v] *= corr[v];

      // --- P tile: C-layout -> LDS (A-fragment layout read below) ---
      const int r  = mt * 16 + ((lane >> 4) << 3);
      const int cc = lane & 15;
#pragma unroll
      for (int v = 0; v < 8; ++v) {
        plds[wave][r + v][cc]      = (bf16)p0[v];
        plds[wave][r + v][cc + 16] = (bf16)p1[v];
      }
    }
    asm volatile("s_wait_dscnt 0" ::: "memory");  // wave-local RAW on LDS
    const v16bf pa0 = load_frag(&plds[wave][lane & 15][0], 0, lane);
    const v16bf pa1 = load_frag(&plds[wave][16 + (lane & 15)][0], 0, lane);

    // --- O += P V : V fragments shared by both M tiles ---
#pragma unroll
    for (int t = 0; t < 4; ++t) {
      const bf16* vrow = vtbase + (size_t)(t * 16 + (lane & 15)) * SEQ;
      const v16bf vb = load_frag(vrow, j0, lane);
      acc[0][t] = wmma_bf16(pa0, vb, acc[0][t]);
      acc[1][t] = wmma_bf16(pa1, vb, acc[1][t]);
    }
  }

  // epilogue: normalize, store bf16 [b, n, h*64+d] for the projection GEMM
#pragma unroll
  for (int mt = 0; mt < 2; ++mt) {
    const int mb = i0 + mt * 16 + ((lane >> 4) << 3);
#pragma unroll
    for (int v = 0; v < 8; ++v) {
      const float inv = 1.f / lrun[mt][v];
      const int row = mb + v;
#pragma unroll
      for (int t = 0; t < 4; ++t) {
        const int d = t * 16 + (lane & 15);
        aows[(size_t)(b * SEQ + row) * DIM + h * HD + d] =
            (bf16)(acc[mt][t][v] * inv);
      }
    }
  }
}

// ---------------------------------------------------------------------------
// Projection GEMM: [8192 x 768] @ [768 x 768]^T + bias -> d_out[0]
// Same 32x64 wave tiling as qkv_gemm.
// ---------------------------------------------------------------------------
__global__ __launch_bounds__(128) void proj_gemm(
    const bf16* __restrict__ ab, const bf16* __restrict__ wb,
    const float* __restrict__ bias, float* __restrict__ out) {
  const int lane = threadIdx.x & 31;
  const int wave = threadIdx.x >> 5;
  const int m0 = blockIdx.x * 64 + (wave & 1) * 32;
  const int n0 = blockIdx.y * 128 + (wave >> 1) * 64;

  const bf16* aptr0 = ab + (size_t)(m0 + (lane & 15)) * DIM;
  const bf16* aptr1 = aptr0 + (size_t)16 * DIM;
  const bf16* bptr[4];
#pragma unroll
  for (int t = 0; t < 4; ++t)
    bptr[t] = wb + (size_t)(n0 + t * 16 + (lane & 15)) * DIM;

  v8f acc[2][4];
#pragma unroll
  for (int mt = 0; mt < 2; ++mt)
#pragma unroll
    for (int t = 0; t < 4; ++t) { v8f z = {}; acc[mt][t] = z; }

  for (int k = 0; k < DIM; k += 32) {
    v16bf a0 = load_frag(aptr0, k, lane);
    v16bf a1 = load_frag(aptr1, k, lane);
#pragma unroll
    for (int t = 0; t < 4; ++t) {
      v16bf bfr = load_frag(bptr[t], k, lane);
      acc[0][t] = wmma_bf16(a0, bfr, acc[0][t]);
      acc[1][t] = wmma_bf16(a1, bfr, acc[1][t]);
    }
  }

#pragma unroll
  for (int mt = 0; mt < 2; ++mt) {
    const int mbase = m0 + mt * 16 + ((lane >> 4) << 3);
#pragma unroll
    for (int t = 0; t < 4; ++t) {
      const int d = n0 + t * 16 + (lane & 15);
      const float bb = bias[d];
#pragma unroll
      for (int v = 0; v < 8; ++v)
        __builtin_nontemporal_store(acc[mt][t][v] + bb,
                                    &out[(size_t)(mbase + v) * DIM + d]);
    }
  }
}

// ---------------------------------------------------------------------------
// Host side
// ---------------------------------------------------------------------------
extern "C" void kernel_launch(void* const* d_in, const int* in_sizes, int n_in,
                              void* d_out, int out_size, void* d_ws, size_t ws_size,
                              hipStream_t stream) {
  const float* x     = (const float*)d_in[0];  // [4,2048,768]
  const float* Wqkv  = (const float*)d_in[1];  // [2304,768]
  const float* Wproj = (const float*)d_in[2];  // [768,768]
  const float* bproj = (const float*)d_in[3];  // [768]
  float* out = (float*)d_out;                  // [out | q | k | v]

  const int n_x  = MTOT * DIM;        // 6291456
  const int n_wq = QKVN * DIM;        // 1769472
  const int n_wp = DIM * DIM;         // 589824
  const int n_ph = BATCH * HEADS * SEQ * HD;  // 6291456

  bf16* xb   = (bf16*)d_ws;
  bf16* wqb  = xb  + n_x;
  bf16* wpb  = wqb + n_wq;
  bf16* qws  = wpb + n_wp;
  bf16* kws  = qws + n_ph;
  bf16* vtws = kws + n_ph;
  bf16* aows = vtws + n_ph;
  // total: ~33.8M bf16 elements (~68 MB) of workspace

  cvt_f32_bf16<<<4096, 256, 0, stream>>>(x, xb, n_x);
  cvt_f32_bf16<<<2048, 256, 0, stream>>>(Wqkv, wqb, n_wq);
  cvt_f32_bf16<<<1024, 256, 0, stream>>>(Wproj, wpb, n_wp);

  qkv_gemm<<<dim3(MTOT / 64, QKVN / 128), 128, 0, stream>>>(
      xb, wqb, out + SLICE, qws, kws, vtws);

  flash_attn<<<dim3(BATCH * HEADS, SEQ / 256), 256, 0, stream>>>(
      qws, kws, vtws, aows);

  proj_gemm<<<dim3(MTOT / 64, DIM / 128), 128, 0, stream>>>(
      aows, wpb, bproj, out);
}
